// BQQLinear_25589415149866
// MI455X (gfx1250) — matrix operand order repaired and checked by reading, not compiled, not run on hardware
//
#include <hip/hip_runtime.h>
#include <hip/hip_bf16.h>
#include <stdint.h>

// ---------------------------------------------------------------------------
// BQQLinear on gfx1250: fold the binary-quantized bilinear form into a dense
// effective weight, then run one bf16 WMMA GEMM (hi/lo split-precision) with
// CDNA5 async global->LDS staging (ASYNCcnt double buffering).
//   Phase 0: quantize X -> integer q (exact in bf16), scale folded into W
//   Phase 1: build Wt[jm][kn] = s * Weff, split into bf16 hi + bf16 lo
//   Phase 2: out = q * (Whi + Wlo) + bias  via v_wmma_f32_16x16x32_bf16,
//            operands staged with global_load_async_to_lds_b128 + ds_load_b128
// ---------------------------------------------------------------------------

#define KDIM 1024   // k*n = 32*32
#define NOUT 1024   // j*m = 32*32

typedef __attribute__((ext_vector_type(16))) __bf16 v16bf;
typedef __attribute__((ext_vector_type(8)))  float  v8f;

struct alignas(32) Frag32B { uint4 lo, hi; };

__device__ __forceinline__ unsigned short f2bf(float f) {
    union { float f; uint32_t u; } v; v.f = f;
    uint32_t u = v.u;
    if ((u & 0x7fffffffu) > 0x7f800000u)           // NaN: keep quiet
        return (unsigned short)((u >> 16) | 0x40);
    u += 0x7fffu + ((u >> 16) & 1u);               // round-to-nearest-even
    return (unsigned short)(u >> 16);
}

__device__ __forceinline__ float bf2f(unsigned short h) {
    union { uint32_t u; float f; } v; v.u = ((uint32_t)h) << 16;
    return v.f;
}

// CDNA5 async copy: 16B per lane, global -> LDS, tracked by ASYNCcnt.
// GVS mode: mem = SGPR64 base + VGPR32 byte offset; VDST = per-lane LDS addr.
__device__ __forceinline__ void async_b128(uint32_t lds_off, unsigned long long base,
                                           uint32_t goff) {
    asm volatile("global_load_async_to_lds_b128 %0, %1, %2"
                 :: "v"(lds_off), "v"(goff), "s"(base) : "memory");
}

// ---------------------------------------------------------------------------
// Phase 0: activation quantization. q = clip(rint(x/s), -127, 127) as bf16
// (integers |q| <= 127 are exact in bf16; the scale s is folded into W).
// ---------------------------------------------------------------------------
__global__ void quant_x_kernel(const float* __restrict__ x,
                               const float* __restrict__ act_scale,
                               unsigned short* __restrict__ q, int total) {
    int i = blockIdx.x * blockDim.x + threadIdx.x;
    if (i >= total) return;
    float s  = fmaxf(fabsf(act_scale[0]), 1e-8f);
    float qv = rintf(x[i] / s);
    qv = fminf(fmaxf(qv, -127.0f), 127.0f);
    q[i] = f2bf(qv);
}

// ---------------------------------------------------------------------------
// Phase 1: build the folded effective weight, transposed: Wt[j*32+m][k*32+n].
// One block per (j,k) pair; 256 threads; p=2, l=8 folded in registers.
// ---------------------------------------------------------------------------
__global__ void build_w_kernel(const float* __restrict__ Y,   // (2,32,32,32,8)
                               const float* __restrict__ Z,   // (2,32,32,8,32)
                               const float* __restrict__ A,   // (2,32,32,4)
                               const float* __restrict__ act_scale,
                               unsigned short* __restrict__ Whi,
                               unsigned short* __restrict__ Wlo) {
    const int j = blockIdx.x >> 5;
    const int k = blockIdx.x & 31;
    const int t = threadIdx.x;          // 0..255

    __shared__ float sY[2][256];        // [p][m*8+l]
    __shared__ float sZ[2][256];        // [p][l*32+n]
    __shared__ float red[256];
    __shared__ float scales[4];         // Ysc0,Ysc1,Zsc0,Zsc1

    for (int p = 0; p < 2; ++p) {
        size_t base = (((size_t)p * 32 + j) * 32 + k) * 256;
        sY[p][t] = Y[base + t];
        sZ[p][t] = Z[base + t];
    }
    __syncthreads();

    for (int r = 0; r < 4; ++r) {       // mean|Y_p|, mean|Z_p|
        red[t] = (r < 2) ? fabsf(sY[r][t]) : fabsf(sZ[r - 2][t]);
        __syncthreads();
        for (int off = 128; off > 0; off >>= 1) {
            if (t < off) red[t] += red[t + off];
            __syncthreads();
        }
        if (t == 0) scales[r] = red[0] * (1.0f / 256.0f);
        __syncthreads();
    }

    float a0[2], a1[2], a2[2], ysc[2], zsc[2], dsum = 0.0f;
    for (int p = 0; p < 2; ++p) {
        const float* ap = A + ((((size_t)p * 32 + j) * 32 + k) * 4);
        a0[p] = ap[0]; a1[p] = ap[1]; a2[p] = ap[2]; dsum += ap[3];
        ysc[p] = scales[p]; zsc[p] = scales[2 + p];
    }
    const float s = fmaxf(fabsf(act_scale[0]), 1e-8f);

    for (int it = 0; it < 4; ++it) {
        int idx = t + it * 256;         // 0..1023 => (m,n)
        int m = idx >> 5, n = idx & 31;
        float w = dsum;
        for (int p = 0; p < 2; ++p) {
            float smn = 0.0f, ym = 0.0f, zn = 0.0f;
            #pragma unroll
            for (int l = 0; l < 8; ++l) {
                float yv = sY[p][m * 8 + l];
                float zv = sZ[p][l * 32 + n];
                float sy = (yv > 0.0f) ? 1.0f : ((yv < 0.0f) ? -1.0f : 0.0f);
                float sz = (zv > 0.0f) ? 1.0f : ((zv < 0.0f) ? -1.0f : 0.0f);
                smn += sy * sz; ym += sy; zn += sz;
            }
            w += a0[p] * ysc[p] * zsc[p] * smn
               + a1[p] * ysc[p] * ym
               + a2[p] * zsc[p] * zn;
        }
        w *= s;                                         // fold activation scale
        unsigned short hi = f2bf(w);
        float lo = w - bf2f(hi);                        // split-precision residual
        size_t o = ((size_t)(j * 32 + m)) * KDIM + (size_t)(k * 32 + n);
        Whi[o] = hi;
        Wlo[o] = f2bf(lo);
    }
}

// ---------------------------------------------------------------------------
// Phase 2: GEMM  out[M x 1024] = q[M x 1024] * (Whi+Wlo)^T + bias
// Workgroup (8 waves) -> 128x64 output block. Per 32-wide K step:
//   async-stage A(128x32) + Bhi/Blo(64x32) bf16 into double-buffered LDS,
//   each wave computes a 16x64 strip with 8 WMMAs; all fragments are loaded
//   into distinct registers first so ds_loads clause up with partial waits.
// ---------------------------------------------------------------------------
__global__ void __launch_bounds__(256, 1)
gemm_kernel(const unsigned short* __restrict__ q,
            const unsigned short* __restrict__ Whi,
            const unsigned short* __restrict__ Wlo,
            const float* __restrict__ bias,
            float* __restrict__ out, int M) {
    __shared__ __align__(16) unsigned short ldsA [2][128 * 32]; // [row][k]  8KB/stage
    __shared__ __align__(16) unsigned short ldsBh[2][ 64 * 32]; // [n][k]    4KB/stage
    __shared__ __align__(16) unsigned short ldsBl[2][ 64 * 32]; // [n][k]    4KB/stage

    const int tid   = threadIdx.x;
    const int wavei = tid >> 5;
    const int lane  = tid & 31;
    const int half  = lane >> 4;
    const int l16   = lane & 15;

    const int mb    = blockIdx.x >> 4;            // NOUT/64 == 16 N-blocks
    const int nb    = blockIdx.x & 15;
    const int mBase = mb * 128;
    const int nBase = nb * 64;

    const unsigned long long qB  = (unsigned long long)(uintptr_t)q;
    const unsigned long long whB = (unsigned long long)(uintptr_t)Whi;
    const unsigned long long wlB = (unsigned long long)(uintptr_t)Wlo;

    constexpr int STAGES = KDIM / 32;             // 32

    // ---- stage issuer: 4 async b128 per thread (2 for A, 1 Bh, 1 Bl) ----
    auto issue_stage = [&](int s) {
        const int b  = s & 1;
        const int k0 = s * 32;
        #pragma unroll
        for (int c = 0; c < 2; ++c) {             // A: 512 x 16B chunks
            int chunk = tid + c * 256;
            int row   = chunk >> 2;
            int part  = chunk & 3;
            int grow  = min(mBase + row, M - 1);
            uint32_t goff = (uint32_t)(((size_t)grow * KDIM + k0) * 2) + part * 16;
            uint32_t loff = (uint32_t)(uintptr_t)&ldsA[b][row * 32] + part * 16;
            async_b128(loff, qB, goff);
        }
        {                                          // Bhi + Blo: 256 x 16B each
            int n    = tid >> 2;
            int part = tid & 3;
            uint32_t goff = (uint32_t)(((size_t)(nBase + n) * KDIM + k0) * 2) + part * 16;
            async_b128((uint32_t)(uintptr_t)&ldsBh[b][n * 32] + part * 16, whB, goff);
            async_b128((uint32_t)(uintptr_t)&ldsBl[b][n * 32] + part * 16, wlB, goff);
        }
    };

    v8f acc[4] = {v8f{}, v8f{}, v8f{}, v8f{}};

    issue_stage(0);
    for (int s = 0; s < STAGES; ++s) {
        if (s + 1 < STAGES) {
            issue_stage(s + 1);                    // prefetch next stage
            asm volatile("s_wait_asynccnt 4" ::: "memory");  // stage s done (in-order)
        } else {
            asm volatile("s_wait_asynccnt 0" ::: "memory");
        }
        __syncthreads();                           // stage s visible to all waves

        const int b = s & 1;

        // ---- load ALL fragments first (distinct regs -> clause + partial waits)
        Frag32B af;
        const uint4* ap = (const uint4*)&ldsA[b][(wavei * 16 + l16) * 32 + half * 16];
        af.lo = ap[0];                             // K = half*16 + [0..7]
        af.hi = ap[1];                             // K = half*16 + 8 + [0..7]
        v16bf a = __builtin_bit_cast(v16bf, af);

        v16bf bh[4], bl[4];
        #pragma unroll
        for (int nn = 0; nn < 4; ++nn) {
            Frag32B fh, fl;
            const uint4* bph = (const uint4*)&ldsBh[b][(nn * 16 + l16) * 32 + half * 16];
            fh.lo = bph[0]; fh.hi = bph[1];        // K = half*16 + [0..15]
            const uint4* bpl = (const uint4*)&ldsBl[b][(nn * 16 + l16) * 32 + half * 16];
            fl.lo = bpl[0]; fl.hi = bpl[1];
            bh[nn] = __builtin_bit_cast(v16bf, fh);
            bl[nn] = __builtin_bit_cast(v16bf, fl);
        }

        // ---- hi pass then lo pass: consecutive WMMAs use distinct accumulators
        #pragma unroll
        for (int nn = 0; nn < 4; ++nn)
            acc[nn] = __builtin_amdgcn_wmma_f32_16x16x32_bf16(
                false, a, false, bh[nn], (short)0, acc[nn], false, false);
        #pragma unroll
        for (int nn = 0; nn < 4; ++nn)
            acc[nn] = __builtin_amdgcn_wmma_f32_16x16x32_bf16(
                false, a, false, bl[nn], (short)0, acc[nn], false, false);

        __syncthreads();                           // done reading buf b
    }

    // C/D layout: VGPR r holds (M = r + half*8, N = lane&15)
    const int n0 = nBase + l16;
    const float b0 = bias[n0], b1 = bias[n0 + 16], b2 = bias[n0 + 32], b3 = bias[n0 + 48];
    #pragma unroll
    for (int r = 0; r < 8; ++r) {
        int mrow = mBase + wavei * 16 + half * 8 + r;
        if (mrow < M) {
            float* orow = out + (size_t)mrow * NOUT;
            orow[n0]      = acc[0][r] + b0;
            orow[n0 + 16] = acc[1][r] + b1;
            orow[n0 + 32] = acc[2][r] + b2;
            orow[n0 + 48] = acc[3][r] + b3;
        }
    }
}

// ---------------------------------------------------------------------------
extern "C" void kernel_launch(void* const* d_in, const int* in_sizes, int n_in,
                              void* d_out, int out_size, void* d_ws, size_t ws_size,
                              hipStream_t stream) {
    const float* x    = (const float*)d_in[0];   // (2,1024,1024) f32
    const float* Y    = (const float*)d_in[1];   // (2,32,32,32,8)
    const float* Z    = (const float*)d_in[2];   // (2,32,32,8,32)
    const float* A    = (const float*)d_in[3];   // (2,32,32,4)
    const float* bias = (const float*)d_in[4];   // (1024,)
    const float* act  = (const float*)d_in[5];   // (1,)

    const int M = in_sizes[0] / KDIM;            // 2048 batch rows

    unsigned short* q   = (unsigned short*)d_ws;                       // M*K bf16
    unsigned short* Whi = (unsigned short*)((char*)d_ws + (size_t)M * KDIM * 2);
    unsigned short* Wlo = Whi + (size_t)NOUT * KDIM;
    float* out = (float*)d_out;

    const int total = M * KDIM;
    quant_x_kernel<<<(total + 255) / 256, 256, 0, stream>>>(x, act, q, total);
    build_w_kernel<<<32 * 32, 256, 0, stream>>>(Y, Z, A, act, Whi, Wlo);

    const int mBlocks = (M + 127) / 128;
    gemm_kernel<<<mBlocks * 16, 256, 0, stream>>>(q, Whi, Wlo, bias, out, M);
}
